// RAVNet_7627861918357
// MI455X (gfx1250) — compile-verified, matrix operand order checked by reading, not run on hardware
//
#include <hip/hip_runtime.h>
#include <hip/hip_bf16.h>

typedef __attribute__((ext_vector_type(16))) _Float16 v16h;
typedef __attribute__((ext_vector_type(8)))  _Float16 v8h;
typedef __attribute__((ext_vector_type(8)))  float    v8f;

#define NTHREADS 128
#define CPAD 96

// ---- WMMA f16 16x16x32 operand helpers (wave32) -----------------------------
// A 16x32 f16 layout: lane l -> row m=l&15, half=l>>4; element i (vgpr i/2):
//   vg<4 : K = half*8 + 2*vg + (i&1)        (contiguous K run half*8+0..7)
//   vg>=4: K = 16 + half*8 + 2*(vg-4)+(i&1) (contiguous K run 16+half*8+0..7)
__device__ __forceinline__ int kmap16(int i, int hf) {
  int vg = i >> 1;
  int base = (vg < 4) ? (hf * 8 + 2 * vg) : (16 + hf * 8 + 2 * (vg - 4));
  return base + (i & 1);
}

// A-style fragment from row-major f16 LDS (two 16B contiguous ds loads/lane).
// Also serves as the B loader for any matrix stored TRANSPOSED in LDS:
// B[k][n] = T[n][k]  ->  load_a16(T, n0, k0, ldT, lane).
__device__ __forceinline__ v16h load_a16(const _Float16* S, int row0, int k0,
                                         int ld, int lane) {
  int m = lane & 15, hf = lane >> 4;
  const _Float16* rp = S + (row0 + m) * ld + k0 + hf * 8;
  v8h lo = *(const v8h*)rp;
  v8h hi = *(const v8h*)(rp + 16);
  v16h a;
#pragma unroll
  for (int i = 0; i < 8; ++i) { a[i] = lo[i]; a[i + 8] = hi[i]; }
  return a;
}

// B fragment from pre-packed fragment-major f16 weights in d_ws:
// tile t occupies 512 halves; lane's 16 halves are contiguous (32 B) -> 2x b128
__device__ __forceinline__ v16h load_bw(const _Float16* __restrict__ F,
                                        int tile, int lane) {
  return *(const v16h*)(F + (size_t)tile * 512 + lane * 16);
}

// C/D f32 fragment -> f16 LDS row-major (scatter: 8x ds_store_b16);
// lane l holds col n=l&15, vgpr r -> row r+8*(l>>4)
__device__ __forceinline__ void store_c16(_Float16* D, int row0, int col0,
                                          int ld, v8f c, int lane) {
  int n = col0 + (lane & 15), hf = lane >> 4;
#pragma unroll
  for (int r = 0; r < 8; ++r)
    D[(row0 + hf * 8 + r) * ld + n] = (_Float16)c[r];
}

// C/D f32 fragment -> f16 LDS TRANSPOSED (column-major): lane's 8 row-values
// are contiguous -> one 16B ds_store_b128 per tile.
__device__ __forceinline__ void store_ct16(_Float16* T, int row0, int col0,
                                           int ldT, v8f c, int lane) {
  int n = col0 + (lane & 15), hf = lane >> 4;
  _Float16* p = T + n * ldT + row0 + hf * 8;
  v8h v;
#pragma unroll
  for (int r = 0; r < 8; ++r) v[r] = (_Float16)c[r];
  *(v8h*)p = v;
}

__device__ __forceinline__ v8f wmma16(v16h a, v16h b, v8f c) {
  return __builtin_amdgcn_wmma_f32_16x16x32_f16(false, a, false, b, (short)0, c,
                                                false, false);
}

// ---- weight pre-pack: W (K x N, f32) -> fragment-major f16 tiles ------------
// grid = KT*NT blocks of 32 lanes; tile = kt*NT + nt; OOB -> 0
__global__ void pack_b_kernel(const float* __restrict__ W, int K, int N, int NT,
                              _Float16* __restrict__ out) {
  int tile = blockIdx.x;
  int kt = tile / NT, nt = tile - kt * NT;
  int lane = threadIdx.x;
  int n = nt * 16 + (lane & 15), hf = lane >> 4;
  _Float16* dst = out + (size_t)tile * 512 + lane * 16;
#pragma unroll
  for (int i = 0; i < 16; ++i) {
    int k = kt * 32 + kmap16(i, hf);
    float v = (k < K && n < N) ? W[k * N + n] : 0.0f;
    dst[i] = (_Float16)v;
  }
}

// ---- exclusive prefix sum of counts -> offsets (single 1024-thread block) ---
__global__ void scan_kernel(const int* __restrict__ counts,
                            int* __restrict__ offsets, int B) {
  __shared__ int sums[1024];
  int t = threadIdx.x;
  int per = (B + 1023) >> 10;
  if (per > 8) per = 8;
  int base = t * per;
  int vals[8];
  int local = 0;
#pragma unroll
  for (int i = 0; i < 8; ++i) {
    int v = 0;
    if (i < per) { int idx = base + i; if (idx < B) v = counts[idx]; }
    vals[i] = v; local += v;
  }
  sums[t] = local;
  __syncthreads();
  for (int d = 1; d < 1024; d <<= 1) {
    int v = (t >= d) ? sums[t - d] : 0;
    __syncthreads();
    sums[t] += v;
    __syncthreads();
  }
  int run = sums[t] - local;  // exclusive prefix of this chunk
#pragma unroll
  for (int i = 0; i < 8; ++i) {
    if (i < per) {
      int idx = base + i;
      if (idx < B) offsets[idx] = run;
      run += vals[i];
    }
  }
}

// ---- main: one workgroup (4 waves) per instance group ----------------------
// LDS map (dynamic, 131072 B):
//   [     0, 24576) X  [128][96] f16   (reused as O [128][64] in phase 3)
//   [ 24576, 49152) XP [128][96] f16
//   [ 49152, 65536) Vt [64][128] f16   (V transposed: column-major)
//   [ 65536, 81920) Q  [128][64] f16
//   [ 81920, 98304) K  [128][64] f16
//   [ 98304,131072) P  [128][128] f16
__global__ __launch_bounds__(NTHREADS) void attn_kernel(
    const float* __restrict__ x, const int* __restrict__ counts,
    const int* __restrict__ offsets, const _Float16* __restrict__ fwq,
    const _Float16* __restrict__ fwk, const _Float16* __restrict__ fwv,
    const _Float16* __restrict__ fprior, const _Float16* __restrict__ fwu,
    const float* __restrict__ bq, const float* __restrict__ bk,
    const float* __restrict__ bv, const float* __restrict__ bu,
    float* __restrict__ out) {
  extern __shared__ __align__(16) char smem[];
  _Float16* Xs  = (_Float16*)(smem);
  _Float16* XPs = (_Float16*)(smem + 24576);
  _Float16* Vt  = (_Float16*)(smem + 49152);  // [64][128] transposed
  _Float16* Qs  = (_Float16*)(smem + 65536);
  _Float16* Ks  = (_Float16*)(smem + 81920);
  _Float16* Ps  = (_Float16*)(smem + 98304);
  _Float16* Os  = (_Float16*)(smem);  // overlays X (dead after phase 1)

  int gb   = blockIdx.x;
  int n    = counts[gb];
  int off  = offsets[gb];
  int Mp   = (n + 31) & ~31;   // pad rows to multiple of 32 (K-legal everywhere)
  int Mt   = Mp >> 4;          // 16-row tiles: 2..8
  int tid  = threadIdx.x;
  int lane = tid & 31;
  int wave = tid >> 5;

  // -- phase 0: zero-pad + stage x (f32 -> f16) into LDS, coalesced ----------
  unsigned int* Xu = (unsigned int*)Xs;
  int nwords = Mp * (CPAD / 2);
  for (int i = tid; i < nwords; i += NTHREADS) Xu[i] = 0u;
  __syncthreads();
  const float* xrow = x + (size_t)off * 81;
  int nelems = n * 81;
  for (int e = tid; e < nelems; e += NTHREADS) {
    int r = e / 81;
    int c = e - r * 81;
    Xs[r * CPAD + c] = (_Float16)xrow[e];
  }
  __syncthreads();

  // -- phase 1: Q = X wq + bq, K = X wk + bk, XP = X prior -------------------
  for (int rt = wave; rt < Mt; rt += 4) {
    int r0 = rt * 16;
    v16h ax0 = load_a16(Xs, r0, 0,  CPAD, lane);   // hoisted: invariant per rt
    v16h ax1 = load_a16(Xs, r0, 32, CPAD, lane);
    v16h ax2 = load_a16(Xs, r0, 64, CPAD, lane);
    for (int nt = 0; nt < 4; ++nt) {
      v8f accq = {}; v8f acck = {};
      accq = wmma16(ax0, load_bw(fwq, 0 * 4 + nt, lane), accq);
      accq = wmma16(ax1, load_bw(fwq, 1 * 4 + nt, lane), accq);
      accq = wmma16(ax2, load_bw(fwq, 2 * 4 + nt, lane), accq);
      acck = wmma16(ax0, load_bw(fwk, 0 * 4 + nt, lane), acck);
      acck = wmma16(ax1, load_bw(fwk, 1 * 4 + nt, lane), acck);
      acck = wmma16(ax2, load_bw(fwk, 2 * 4 + nt, lane), acck);
      float biq = bq[nt * 16 + (lane & 15)];
      float bik = bk[nt * 16 + (lane & 15)];
#pragma unroll
      for (int r = 0; r < 8; ++r) { accq[r] += biq; acck[r] += bik; }
      store_c16(Qs, r0, nt * 16, 64, accq, lane);
      store_c16(Ks, r0, nt * 16, 64, acck, lane);
    }
    for (int nt = 0; nt < 6; ++nt) {
      v8f acc = {};
      acc = wmma16(ax0, load_bw(fprior, 0 * 6 + nt, lane), acc);
      acc = wmma16(ax1, load_bw(fprior, 1 * 6 + nt, lane), acc);
      acc = wmma16(ax2, load_bw(fprior, 2 * 6 + nt, lane), acc);
      store_c16(XPs, r0, nt * 16, CPAD, acc, lane);
    }
  }
  __syncthreads();

  // -- phase 2: V = XP wv + bv, stored TRANSPOSED (Vt[dk][row]) --------------
  for (int rt = wave; rt < Mt; rt += 4) {
    int r0 = rt * 16;
    v16h ap0 = load_a16(XPs, r0, 0,  CPAD, lane);
    v16h ap1 = load_a16(XPs, r0, 32, CPAD, lane);
    v16h ap2 = load_a16(XPs, r0, 64, CPAD, lane);
    for (int nt = 0; nt < 4; ++nt) {
      v8f acc = {};
      acc = wmma16(ap0, load_bw(fwv, 0 * 4 + nt, lane), acc);
      acc = wmma16(ap1, load_bw(fwv, 1 * 4 + nt, lane), acc);
      acc = wmma16(ap2, load_bw(fwv, 2 * 4 + nt, lane), acc);
      float biv = bv[nt * 16 + (lane & 15)];
#pragma unroll
      for (int r = 0; r < 8; ++r) acc[r] += biv;
      store_ct16(Vt, r0, nt * 16, 128, acc, lane);  // 1x ds_store_b128
    }
  }
  __syncthreads();

  // -- phase 3: scores -> softmax -> P, O = P V, out = O wu + bu -------------
  int Kt = Mp >> 5;  // 32-wide K steps over probs
  for (int rt = wave; rt < Mt; rt += 4) {
    int r0 = rt * 16;
    v16h aq0 = load_a16(Qs, r0, 0,  64, lane);     // hoisted Q fragments
    v16h aq1 = load_a16(Qs, r0, 32, 64, lane);
    v8f s[8];
#pragma unroll
    for (int ct = 0; ct < 8; ++ct) {
      if (ct < Mt) {  // block-uniform branch: EXEC stays all-ones for WMMA
        v8f acc = {};
        v16h bt0 = load_a16(Ks, ct * 16, 0,  64, lane);  // K^T fragments
        v16h bt1 = load_a16(Ks, ct * 16, 32, 64, lane);
        acc = wmma16(aq0, bt0, acc);
        acc = wmma16(aq1, bt1, acc);
        int col = ct * 16 + (lane & 15);
#pragma unroll
        for (int r = 0; r < 8; ++r)
          acc[r] = (col < n) ? acc[r] * 0.125f : -1e30f;
        s[ct] = acc;
      } else {
        v8f z;
#pragma unroll
        for (int r = 0; r < 8; ++r) z[r] = -1e30f;
        s[ct] = z;
      }
    }
    // softmax over key columns; a row's 16 tile-cols live in one 16-lane half
    float mx[8], sum[8];
#pragma unroll
    for (int r = 0; r < 8; ++r) mx[r] = -1e30f;
#pragma unroll
    for (int ct = 0; ct < 8; ++ct)
#pragma unroll
      for (int r = 0; r < 8; ++r) mx[r] = fmaxf(mx[r], s[ct][r]);
#pragma unroll
    for (int r = 0; r < 8; ++r) {
#pragma unroll
      for (int d = 1; d < 16; d <<= 1)
        mx[r] = fmaxf(mx[r], __shfl_xor(mx[r], d, 16));
      sum[r] = 0.0f;
    }
#pragma unroll
    for (int ct = 0; ct < 8; ++ct)
#pragma unroll
      for (int r = 0; r < 8; ++r) {
        float e = __expf(s[ct][r] - mx[r]);
        s[ct][r] = e;
        sum[r] += e;
      }
#pragma unroll
    for (int r = 0; r < 8; ++r) {
#pragma unroll
      for (int d = 1; d < 16; d <<= 1) sum[r] += __shfl_xor(sum[r], d, 16);
      sum[r] = 1.0f / sum[r];
    }
#pragma unroll
    for (int ct = 0; ct < 8; ++ct) {
      v8f p;
#pragma unroll
      for (int r = 0; r < 8; ++r) p[r] = s[ct][r] * sum[r];
      store_c16(Ps, r0, ct * 16, 128, p, lane);
    }
    // O = P @ V: B fragment = Vt (transposed) via the vectorized loader
    v16h apf[4];
#pragma unroll
    for (int kt = 0; kt < 4; ++kt)
      if (kt < Kt) apf[kt] = load_a16(Ps, r0, kt * 32, 128, lane);
    for (int nt = 0; nt < 4; ++nt) {
      v8f acc = {};
#pragma unroll
      for (int kt = 0; kt < 4; ++kt) {
        if (kt < Kt) {  // block-uniform
          v16h fb = load_a16(Vt, nt * 16, kt * 32, 128, lane);  // B[k][n]=Vt[n][k]
          acc = wmma16(apf[kt], fb, acc);
        }
      }
      store_c16(Os, r0, nt * 16, 64, acc, lane);
    }
    // out = O @ wu + bu -> ragged global scatter (f32)
    v16h ao0 = load_a16(Os, r0, 0,  64, lane);
    v16h ao1 = load_a16(Os, r0, 32, 64, lane);
    int hf = lane >> 4, nn = lane & 15;
    for (int ct2 = 0; ct2 < 6; ++ct2) {
      v8f acc = {};
      acc = wmma16(ao0, load_bw(fwu, 0 * 6 + ct2, lane), acc);
      acc = wmma16(ao1, load_bw(fwu, 1 * 6 + ct2, lane), acc);
      int col = ct2 * 16 + nn;
      if (col < 81) {
        float bias = bu[col];
#pragma unroll
        for (int r = 0; r < 8; ++r) {
          int grow = r0 + hf * 8 + r;
          if (grow < n)
            out[(size_t)(off + grow) * 81 + col] = acc[r] + bias;
        }
      }
    }
  }
}

extern "C" void kernel_launch(void* const* d_in, const int* in_sizes, int n_in,
                              void* d_out, int out_size, void* d_ws,
                              size_t ws_size, hipStream_t stream) {
  const float* x      = (const float*)d_in[0];
  const int*   counts = (const int*)d_in[1];
  const float* prior  = (const float*)d_in[2];
  const float* wq     = (const float*)d_in[3];
  const float* bq     = (const float*)d_in[4];
  const float* wk     = (const float*)d_in[5];
  const float* bk     = (const float*)d_in[6];
  const float* wv     = (const float*)d_in[7];
  const float* bv     = (const float*)d_in[8];
  const float* wu     = (const float*)d_in[9];
  const float* bu     = (const float*)d_in[10];
  int B = in_sizes[1];

  // d_ws layout: [0,16K) offsets; then packed f16 weight fragments (1KB/tile)
  char* ws = (char*)d_ws;
  int*      offsets = (int*)ws;
  _Float16* fwq     = (_Float16*)(ws + 16384);           // 3x4 tiles  = 12 KB
  _Float16* fwk     = (_Float16*)(ws + 16384 + 12288);   // 3x4 tiles  = 12 KB
  _Float16* fwv     = (_Float16*)(ws + 16384 + 24576);   // 3x4 tiles  = 12 KB
  _Float16* fprior  = (_Float16*)(ws + 16384 + 36864);   // 3x6 tiles  = 18 KB
  _Float16* fwu     = (_Float16*)(ws + 16384 + 55296);   // 2x6 tiles  = 12 KB

  scan_kernel<<<1, 1024, 0, stream>>>(counts, offsets, B);
  pack_b_kernel<<<3 * 4, 32, 0, stream>>>(wq,    81, 64, 4, fwq);
  pack_b_kernel<<<3 * 4, 32, 0, stream>>>(wk,    81, 64, 4, fwk);
  pack_b_kernel<<<3 * 4, 32, 0, stream>>>(wv,    81, 64, 4, fwv);
  pack_b_kernel<<<3 * 6, 32, 0, stream>>>(prior, 81, 81, 6, fprior);
  pack_b_kernel<<<2 * 6, 32, 0, stream>>>(wu,    64, 81, 6, fwu);

  size_t lds_bytes = 131072;  // 128 KB of the 320 KB/WGP
  attn_kernel<<<B, NTHREADS, lds_bytes, stream>>>(
      x, counts, offsets, fwq, fwk, fwv, fprior, fwu, bq, bk, bv, bu,
      (float*)d_out);
}
